// MHAP_20005957665165
// MI455X (gfx1250) — matrix operand order repaired and checked by reading, not compiled, not run on hardware
//
#include <hip/hip_runtime.h>
#include <stdint.h>

typedef __attribute__((ext_vector_type(16))) __bf16   v16bf;
typedef __attribute__((ext_vector_type(8)))  __bf16   v8bf;
typedef __attribute__((ext_vector_type(8)))  float    v8f;
typedef __attribute__((ext_vector_type(4)))  int      v4i;
typedef __attribute__((ext_vector_type(4)))  uint16_t v4u16;

// LDS (addrspace 3) qualified views — keep all panel addressing in AS3 so the
// backend emits ds_load/ds_store with 32-bit addresses (never flat_*).
typedef __attribute__((address_space(3))) __bf16    lds_bf16;
typedef __attribute__((address_space(3))) uint16_t  lds_u16;
typedef __attribute__((address_space(3))) v4u16     lds_v4u16;
typedef __attribute__((address_space(3))) const v8bf lds_cv8bf;
typedef __attribute__((address_space(3))) v4i       lds_v4i;
typedef __attribute__((address_space(1))) const __bf16 glb_bf16;

#define B_  64
#define T_  512
#define D_  2048
#define H_  8
#define DR_ 512

// scores-kernel LDS geometry
#define MTILE   64
#define A_STR   (D_ + 8)        // 2056 bf16/row: +16B pad -> conflict-free frag reads
#define KC      64              // W-panel K chunk staged per pipeline stage
#define B_STR   (KC + 8)        // 72 elems = 144B row stride (16B aligned, no conflicts)
#define NTILE   128             // N columns per iteration (8 blocks of 16)
#define NCHUNK  (D_ / KC)       // 32
#define BBUFSZ  (NTILE * B_STR) // elems per B panel buffer

__device__ __forceinline__ uint16_t f2bf_bits(float f) {
    union { float f; uint32_t u; } v; v.f = f;
    uint32_t u = v.u;
    return (uint16_t)((u + 0x7FFFu + ((u >> 16) & 1u)) >> 16);   // RNE
}
__device__ __forceinline__ __bf16 f2bf(float f) {
    union { uint16_t s; __bf16 b; } o; o.s = f2bf_bits(f);
    return o.b;
}

// ---- CDNA5 async global->LDS copy (ASYNCcnt path), 16B per call -------------
__device__ __forceinline__ void async_cp16(lds_bf16* lds, glb_bf16* g) {
#if __has_builtin(__builtin_amdgcn_global_load_async_to_lds_b128)
    __builtin_amdgcn_global_load_async_to_lds_b128(
        (__attribute__((address_space(1))) v4i*)(g), (lds_v4i*)(lds), 0, 0);
#else
    uint32_t l = (uint32_t)(uintptr_t)lds;
    uint64_t ga = (uint64_t)(uintptr_t)g;
    asm volatile("global_load_async_to_lds_b128 %0, %1, off"
                 :: "v"(l), "v"(ga) : "memory");
#endif
}

template <int N>
__device__ __forceinline__ void wait_async() {
#if __has_builtin(__builtin_amdgcn_s_wait_asynccnt)
    __builtin_amdgcn_s_wait_asynccnt((short)N);
#else
    asm volatile("s_wait_asynccnt %0" :: "i"(N) : "memory");
#endif
}

// ---------------------------------------------------------------------------
// W1 [H, D, DR] f32  ->  wbf [H*DR, D] bf16 (transposed, K-contiguous rows)
// ---------------------------------------------------------------------------
__global__ __launch_bounds__(256) void sap_wtrans(const float* __restrict__ W1,
                                                  __bf16* __restrict__ wbf) {
    __shared__ float tile[32][33];
    const int h = blockIdx.z, d0 = blockIdx.x * 32, r0 = blockIdx.y * 32;
    const int tx = threadIdx.x, ty = threadIdx.y;
    const float* src = W1 + (size_t)h * D_ * DR_;
#pragma unroll
    for (int j = 0; j < 4; ++j)
        tile[ty + 8 * j][tx] = src[(size_t)(d0 + ty + 8 * j) * DR_ + r0 + tx];
    __syncthreads();
#pragma unroll
    for (int j = 0; j < 4; ++j)
        wbf[(size_t)(h * DR_ + r0 + ty + 8 * j) * D_ + d0 + tx] =
            f2bf(tile[tx][ty + 8 * j]);
}

// ---------------------------------------------------------------------------
// Fused scores: scores[b,h,t] = sum_r tanh(x@W1 + b1) * W2  (+ b2 at flush)
// 64 token rows x full K resident in LDS; async double-buffered W panels.
// ---------------------------------------------------------------------------
__global__ __launch_bounds__(256) void sap_scores(
        const float* __restrict__ x, const __bf16* __restrict__ wbf,
        const float* __restrict__ b1g, const float* __restrict__ W2,
        const float* __restrict__ b2, float* __restrict__ scores) {
    extern __shared__ __align__(16) char smem[];
    // one generic->AS3 cast each, at function scope; all later addressing is AS3
    lds_bf16* Ald = (lds_bf16*)smem;                                  // MTILE x A_STR
    lds_bf16* Bt  = (lds_bf16*)(smem + (size_t)MTILE * A_STR * 2);    // 2 x BBUFSZ
    float*    scL = (float*)(smem + (size_t)MTILE * A_STR * 2
                                  + (size_t)2 * BBUFSZ * 2);
    glb_bf16* wg  = (glb_bf16*)wbf;

    const int tid = threadIdx.x;
    const int g   = blockIdx.x;
    const int b   = g >> 3;            // T_/MTILE = 8 tiles per batch
    const int t0  = (g & 7) * MTILE;

    // stage A: x[b, t0:t0+64, :] f32 -> bf16 LDS (x read from HBM exactly once)
    const float4* src = (const float4*)(x + ((size_t)(b * T_ + t0)) * D_);
    for (int f = tid; f < MTILE * (D_ / 4); f += 256) {
        float4 v = src[f];
        int row = f >> 9;              // D_/4 = 512 float4 per row
        int c4  = f & 511;
        v4u16 p;
        p.x = f2bf_bits(v.x); p.y = f2bf_bits(v.y);
        p.z = f2bf_bits(v.z); p.w = f2bf_bits(v.w);
        *(lds_v4u16*)((lds_u16*)Ald + (size_t)row * A_STR + c4 * 4) = p;
    }

    const int lane = tid & 31;
    const int wv   = tid >> 5;         // 8 waves
    const int l16  = lane & 15;
    const int hl   = lane >> 4;        // lane half
    const int mb   = wv & 3;           // m-block (16 rows), waves 0-3 / 4-7 pair up
    const int nbq  = (wv >> 2) * 4;    // first of four n-blocks

    lds_bf16* Arow = Ald + (size_t)(mb * 16 + l16) * A_STR;

    for (int nt = 0; nt < (H_ * DR_) / NTILE; ++nt) {
        const int nbase = nt * NTILE;
        const int hd = nbase >> 9;          // head
        const int rb = nbase & (DR_ - 1);   // r offset within head
        if (rb == 0 && tid < MTILE) scL[tid] = 0.0f;
        __syncthreads();                    // A-stage / zero / prev-read fence

        v8f acc[4] = {};

        // prefetch chunk 0 (4 async b128 per thread)
#pragma unroll
        for (int c = tid; c < NTILE * (KC / 8); c += 256) {
            int row = c >> 3, cc = c & 7;
            async_cp16(Bt + (unsigned)(row * B_STR + cc * 8),
                       wg + (size_t)(nbase + row) * D_ + cc * 8);
        }

        for (int ic = 0; ic < NCHUNK; ++ic) {
            const unsigned curo = (unsigned)(ic & 1) * BBUFSZ;     // int select, not ptr
            if (ic + 1 < NCHUNK) {          // prefetch next chunk into other buffer
                const unsigned nxto = curo ^ BBUFSZ;
#pragma unroll
                for (int c = tid; c < NTILE * (KC / 8); c += 256) {
                    int row = c >> 3, cc = c & 7;
                    async_cp16(Bt + nxto + (unsigned)(row * B_STR + cc * 8),
                               wg + (size_t)(nbase + row) * D_ + (ic + 1) * KC + cc * 8);
                }
                wait_async<4>();            // in-order: drains current chunk's 4 loads
            } else {
                wait_async<0>();
            }
            __syncthreads();

            lds_bf16* Bb = Bt + curo + (unsigned)(l16 * B_STR + 16 * hl);
#pragma unroll
            for (int kk = 0; kk < KC; kk += 32) {
                // A fragment: half 0 -> K 0-15, half 1 -> K 16-31
                v8bf alo = *(lds_cv8bf*)(Arow + ic * KC + kk + 16 * hl);
                v8bf ahi = *(lds_cv8bf*)(Arow + ic * KC + kk + 16 * hl + 8);
                v16bf af;
#pragma unroll
                for (int i = 0; i < 8; ++i) { af[i] = alo[i]; af[i + 8] = ahi[i]; }
#pragma unroll
                for (int a = 0; a < 4; ++a) {   // B fragments: lane = N, K = 16*half + e
                    lds_bf16* br = Bb + (unsigned)((nbq + a) * 16 * B_STR + kk);
                    v8bf blo = *(lds_cv8bf*)br;
                    v8bf bhi = *(lds_cv8bf*)(br + 8);
                    v16bf bv;
#pragma unroll
                    for (int i = 0; i < 8; ++i) { bv[i] = blo[i]; bv[i + 8] = bhi[i]; }
                    acc[a] = __builtin_amdgcn_wmma_f32_16x16x32_bf16(
                                 false, af, false, bv, (short)0, acc[a], false, false);
                }
            }
            __syncthreads();                // readers done before buffer reuse
        }

        // epilogue: tanh + W2 dot, reduce the 16 N-lanes, accumulate per-row score
#pragma unroll
        for (int a = 0; a < 4; ++a) {
            int r = rb + (nbq + a) * 16 + l16;
            float w2v = W2[hd * DR_ + r];
            float bbv = b1g[hd * DR_ + r];
#pragma unroll
            for (int i = 0; i < 8; ++i) {   // C layout: M = i + 8*half, N = lane&15
                float v = tanhf(acc[a][i] + bbv) * w2v;
                v += __shfl_xor(v, 1, 32);
                v += __shfl_xor(v, 2, 32);
                v += __shfl_xor(v, 4, 32);
                v += __shfl_xor(v, 8, 32);
                if (l16 == 0) atomicAdd(&scL[mb * 16 + hl * 8 + i], v);
            }
        }

        if (rb == DR_ - NTILE) {            // last tile of this head: flush
            __syncthreads();
            if (tid < MTILE)
                scores[(size_t)(b * H_ + hd) * T_ + t0 + tid] = scL[tid] + b2[hd];
        }
    }
}

// ---------------------------------------------------------------------------
// softmax over T per (b,h)
// ---------------------------------------------------------------------------
__global__ __launch_bounds__(256) void sap_softmax(const float* __restrict__ scores,
                                                   float* __restrict__ attn) {
    __shared__ float red[256];
    const int g = blockIdx.x, tid = threadIdx.x;
    const float* s = scores + (size_t)g * T_;
    float v0 = s[tid], v1 = s[tid + 256];
    red[tid] = fmaxf(v0, v1); __syncthreads();
    for (int o = 128; o > 0; o >>= 1) {
        if (tid < o) red[tid] = fmaxf(red[tid], red[tid + o]);
        __syncthreads();
    }
    float mx = red[0];
    __syncthreads();
    float e0 = __expf(v0 - mx), e1 = __expf(v1 - mx);
    red[tid] = e0 + e1; __syncthreads();
    for (int o = 128; o > 0; o >>= 1) {
        if (tid < o) red[tid] += red[tid + o];
        __syncthreads();
    }
    float inv = 1.0f / red[0];
    attn[(size_t)g * T_ + tid]       = e0 * inv;
    attn[(size_t)g * T_ + tid + 256] = e1 * inv;
}

// ---------------------------------------------------------------------------
// pooled[b,h,d] = sum_t attn[b,h,t] * x[b,t,d]   (f32, bandwidth-bound)
// ---------------------------------------------------------------------------
__global__ __launch_bounds__(256) void sap_pool(const float* __restrict__ x,
                                                const float* __restrict__ attn,
                                                float* __restrict__ out) {
    __shared__ float at[H_ * T_];     // 16 KB
    const int b = blockIdx.y;
    const int d = blockIdx.x * 256 + threadIdx.x;
    for (int i = threadIdx.x; i < H_ * T_; i += 256)
        at[i] = attn[(size_t)b * H_ * T_ + i];
    __syncthreads();
    float acc[H_];
#pragma unroll
    for (int h = 0; h < H_; ++h) acc[h] = 0.0f;
    const float* xb = x + (size_t)b * T_ * D_ + d;
    for (int t = 0; t < T_; ++t) {
        float xv = xb[(size_t)t * D_];
#pragma unroll
        for (int h = 0; h < H_; ++h) acc[h] += at[h * T_ + t] * xv;
    }
#pragma unroll
    for (int h = 0; h < H_; ++h)
        out[(size_t)b * (H_ * D_) + h * D_ + d] = acc[h];
}

// ---------------------------------------------------------------------------
extern "C" void kernel_launch(void* const* d_in, const int* in_sizes, int n_in,
                              void* d_out, int out_size, void* d_ws, size_t ws_size,
                              hipStream_t stream) {
    const float* x  = (const float*)d_in[0];
    const float* W1 = (const float*)d_in[1];
    const float* b1 = (const float*)d_in[2];
    const float* W2 = (const float*)d_in[3];
    const float* b2 = (const float*)d_in[4];
    float* out = (float*)d_out;
    char* ws = (char*)d_ws;

    __bf16* wbf   = (__bf16*)ws;                                         // 16 MiB
    float* scores = (float*)(ws + (size_t)H_ * DR_ * D_ * 2);            // 1 MiB
    float* attn   = (float*)(ws + (size_t)H_ * DR_ * D_ * 2
                                + (size_t)B_ * H_ * T_ * 4);             // 1 MiB

    sap_wtrans<<<dim3(D_ / 32, DR_ / 32, H_), dim3(32, 8), 0, stream>>>(W1, wbf);

    const size_t sh = (size_t)MTILE * A_STR * 2                 // 263168
                    + (size_t)2 * BBUFSZ * 2                    // +36864
                    + (size_t)MTILE * 4;                        // +256 = 300288 B
    (void)hipFuncSetAttribute((const void*)sap_scores,
                              hipFuncAttributeMaxDynamicSharedMemorySize, (int)sh);
    sap_scores<<<dim3((B_ * T_) / MTILE), dim3(256), sh, stream>>>(
        x, wbf, b1, W2, b2, scores);

    sap_softmax<<<dim3(B_ * H_), dim3(256), 0, stream>>>(scores, attn);
    sap_pool<<<dim3(D_ / 256, B_), dim3(256), 0, stream>>>(x, attn, out);
}
